// MaskedScaledDotProductAttention_45956150067298
// MI455X (gfx1250) — matrix operand order repaired and checked by reading, not compiled, not run on hardware
//
#include <hip/hip_runtime.h>
#include <hip/hip_bf16.h>

typedef __bf16 bf16_t;
typedef bf16_t v4bf  __attribute__((ext_vector_type(4)));
typedef bf16_t v8bf  __attribute__((ext_vector_type(8)));
typedef bf16_t v16bf __attribute__((ext_vector_type(16)));
typedef float  v8f   __attribute__((ext_vector_type(8)));

#define BATCH 32
#define SEQ   2048
#define DHEAD 64
#define QT    64      // q rows per workgroup
#define KT    32      // key rows per tile
#define NWAVE 4
#define L2E   1.44269504088896340736f

// hardware f32 -> bf16 convert (v_cvt_pk_bf16_f32 path)
__device__ __forceinline__ bf16_t cvt_bf16(float f) { return (bf16_t)f; }

// raw v_exp_f32: args are <= 0 here; -huge flushes to 0 (wanted), no libm fixup
__device__ __forceinline__ float exp2_raw(float f) { return __builtin_amdgcn_exp2f(f); }

__device__ __forceinline__ v16bf cat8(v8bf lo, v8bf hi) {
  v16bf r;
#pragma unroll
  for (int i = 0; i < 8; ++i) { r[i] = lo[i]; r[i + 8] = hi[i]; }
  return r;
}

__device__ __forceinline__ v8bf lds8(const bf16_t* p) { return *(const v8bf*)p; }

__global__ __launch_bounds__(128) void fa_fwd_causal(const float* __restrict__ Q,
                                                     const float* __restrict__ K,
                                                     const float* __restrict__ V,
                                                     float* __restrict__ O) {
  const int qtile = blockIdx.x;
  const int b     = blockIdx.y;
  const int q0    = qtile * QT;
  const int tid   = threadIdx.x;
  const int wave  = tid >> 5;
  const int lane  = tid & 31;
  const int ln    = lane & 15;   // col (B/C/D frag) or row (A frag)
  const int lh    = lane >> 4;   // lane half
  const int ah    = lh * 8;      // A-fragment K offset for this half
  const int bh    = lh * 16;     // unused; B fragments use ah as well
  const int wrow0 = q0 + wave * 16;   // first q row owned by this wave

  __shared__ alignas(16) bf16_t sK [KT][DHEAD + 8];      // K tile, row-major
  __shared__ alignas(16) bf16_t sVt[DHEAD][KT + 8];      // V tile, transposed
  __shared__ alignas(16) bf16_t sP [NWAVE][16][KT + 8];  // per-wave P staging

  const float* Qb = Q + (size_t)b * SEQ * DHEAD;
  const float* Kb = K + (size_t)b * SEQ * DHEAD;
  const float* Vb = V + (size_t)b * SEQ * DHEAD;
  float*       Ob = O + (size_t)b * SEQ * DHEAD;

  // per-thread staging coordinates (8 float4 slots: 4 K + 4 V)
  int srow4[4], scol4[4];
#pragma unroll
  for (int i = 0; i < 4; ++i) {
    const int idx = tid + i * 128;     // 0..511 float4 slots
    srow4[i] = idx >> 4;               // 0..31
    scol4[i] = (idx & 15) * 4;         // 0..60
  }

  // ---- Q fragments (A layout), 1/sqrt(d)=0.125 folded into the convert ----
  const int qrow = wrow0 + ln;
  v16bf qfrag[2];
#pragma unroll
  for (int c = 0; c < 2; ++c) {
    const float* qp = Qb + (size_t)qrow * DHEAD + c * 32;
    const float4 f0 = *(const float4*)(qp + ah);
    const float4 f1 = *(const float4*)(qp + ah + 4);
    const float4 g0 = *(const float4*)(qp + 16 + ah);
    const float4 g1 = *(const float4*)(qp + 16 + ah + 4);
    const float tmp[16] = {f0.x, f0.y, f0.z, f0.w, f1.x, f1.y, f1.z, f1.w,
                           g0.x, g0.y, g0.z, g0.w, g1.x, g1.y, g1.z, g1.w};
    v16bf f;
#pragma unroll
    for (int i = 0; i < 16; ++i) f[i] = cvt_bf16(tmp[i] * 0.125f);
    qfrag[c] = f;
  }

  // all-ones B fragment: one WMMA computes all 16 row-sums of P
  v16bf ones;
#pragma unroll
  for (int i = 0; i < 16; ++i) ones[i] = cvt_bf16(1.0f);

  // ---- flash-attention state ----
  v8f o[4];
#pragma unroll
  for (int j = 0; j < 4; ++j)
#pragma unroll
    for (int r = 0; r < 8; ++r) o[j][r] = 0.0f;
  float m[8], l[8];
#pragma unroll
  for (int r = 0; r < 8; ++r) { m[r] = -1e30f; l[r] = 0.0f; }

  const int nkt = (q0 + QT) / KT;  // causal bound: only tiles with k0 <= q0+63

  // ---- software pipeline: tile t+1 loads fly during tile t compute ----
  float4 kreg[4], vreg[4];
#pragma unroll
  for (int i = 0; i < 4; ++i) {   // issue tile-0 loads back-to-back
    kreg[i] = *(const float4*)(Kb + (size_t)srow4[i] * DHEAD + scol4[i]);
    vreg[i] = *(const float4*)(Vb + (size_t)srow4[i] * DHEAD + scol4[i]);
  }

  for (int t = 0; t < nkt; ++t) {
    const int k0 = t * KT;
    __syncthreads();                       // LDS free from previous compute
    // ---- convert register-buffered tile t into LDS (K row-major, V^T) ----
#pragma unroll
    for (int i = 0; i < 4; ++i) {
      const int r  = srow4[i];
      const int c4 = scol4[i];
      const v4bf kp4 = {cvt_bf16(kreg[i].x), cvt_bf16(kreg[i].y),
                        cvt_bf16(kreg[i].z), cvt_bf16(kreg[i].w)};
      *(v4bf*)&sK[r][c4] = kp4;            // one ds_store_b64
      sVt[c4 + 0][r] = cvt_bf16(vreg[i].x); sVt[c4 + 1][r] = cvt_bf16(vreg[i].y);
      sVt[c4 + 2][r] = cvt_bf16(vreg[i].z); sVt[c4 + 3][r] = cvt_bf16(vreg[i].w);
    }
    // ---- issue tile t+1 loads; they overlap all of tile t's compute ----
    if (t + 1 < nkt) {
      const int kn = k0 + KT;
#pragma unroll
      for (int i = 0; i < 4; ++i) {
        kreg[i] = *(const float4*)(Kb + (size_t)(kn + srow4[i]) * DHEAD + scol4[i]);
        vreg[i] = *(const float4*)(Vb + (size_t)(kn + srow4[i]) * DHEAD + scol4[i]);
      }
    }
    if (t + 2 < nkt) {   // cache hint two tiles ahead (global_prefetch_b8)
      __builtin_prefetch(Kb + (size_t)(k0 + 2 * KT) * DHEAD + tid * 16, 0, 0);
      __builtin_prefetch(Vb + (size_t)(k0 + 2 * KT) * DHEAD + tid * 16, 0, 0);
    }
    __syncthreads();                       // LDS tile t visible to all waves

    // wave-uniform skip: tile entirely in the causal future of this wave's rows
    if (k0 > wrow0 + 15) continue;

    // ---- S = Q K^T : two 16x16 subtiles, K-depth 64 via two WMMAs each ----
    v8f s[2];
#pragma unroll
    for (int sub = 0; sub < 2; ++sub) {
      v8f acc;
#pragma unroll
      for (int r = 0; r < 8; ++r) acc[r] = 0.0f;
#pragma unroll
      for (int c = 0; c < 2; ++c) {
        // B[k][n] = Ktile[sub*16+n][32c+k]: two 8-bf16 runs per lane (k = 8h.., 16+8h..)
        const bf16_t* kp = &sK[sub * 16 + ln][c * 32 + ah];
        const v16bf bfrag = cat8(lds8(kp), lds8(kp + 16));
        acc = __builtin_amdgcn_wmma_f32_16x16x32_bf16(false, qfrag[c], false, bfrag,
                                                      (short)0, acc, false, false);
      }
      s[sub] = acc;
    }

    // causal mask, only on diagonal tiles (wave-uniform test)
    if (k0 + KT - 1 > wrow0) {
#pragma unroll
      for (int sub = 0; sub < 2; ++sub) {
        const int colg = k0 + sub * 16 + ln;
#pragma unroll
        for (int r = 0; r < 8; ++r) {
          const int rowg = wrow0 + r + ah;
          if (colg > rowg) s[sub][r] = -1e30f;
        }
      }
    }

    // ---- online softmax: row max via width-16 shuffles ----
    float sc[8];
#pragma unroll
    for (int r = 0; r < 8; ++r) {
      float v = fmaxf(s[0][r], s[1][r]);
#pragma unroll
      for (int off = 1; off < 16; off <<= 1)
        v = fmaxf(v, __shfl_xor(v, off, 16));
      const float nm = fmaxf(m[r], v);
      sc[r] = exp2_raw((m[r] - nm) * L2E);
      m[r] = nm;
      s[0][r] = exp2_raw((s[0][r] - nm) * L2E);
      s[1][r] = exp2_raw((s[1][r] - nm) * L2E);
#pragma unroll
      for (int j = 0; j < 4; ++j) o[j][r] *= sc[r];
    }

    // ---- convert P to bf16 via per-wave LDS (acc layout -> A layout) ----
#pragma unroll
    for (int sub = 0; sub < 2; ++sub)
#pragma unroll
      for (int r = 0; r < 8; ++r)
        sP[wave][r + ah][sub * 16 + ln] = cvt_bf16(s[sub][r]);
    // same-wave LDS store->load: DS pipe is in-order, but block compiler
    // reordering and drain DScnt explicitly.
    asm volatile("s_wait_dscnt 0x0" ::: "memory");

    const bf16_t* pp = &sP[wave][ln][0];
    const v16bf pfrag = cat8(lds8(pp + ah), lds8(pp + 16 + ah));

    // ---- row sums of P via WMMA against all-ones B (replaces 32 bpermutes) ----
    v8f srow;
#pragma unroll
    for (int r = 0; r < 8; ++r) srow[r] = 0.0f;
    srow = __builtin_amdgcn_wmma_f32_16x16x32_bf16(false, pfrag, false, ones,
                                                   (short)0, srow, false, false);

    // ---- O += P V ----
#pragma unroll
    for (int j = 0; j < 4; ++j) {
      // B[k][n] = Vtile[k][16j+n] = sVt[16j+n][k]: two 8-bf16 runs per lane (k = 8h.., 16+8h..)
      const bf16_t* vp = &sVt[j * 16 + ln][ah];
      const v16bf vfrag = cat8(lds8(vp), lds8(vp + 16));
      o[j] = __builtin_amdgcn_wmma_f32_16x16x32_bf16(false, pfrag, false, vfrag,
                                                     (short)0, o[j], false, false);
    }

#pragma unroll
    for (int r = 0; r < 8; ++r) l[r] = l[r] * sc[r] + srow[r];
  }

  // ---- epilogue: normalize and store (fp32) ----
#pragma unroll
  for (int r = 0; r < 8; ++r) {
    const float inv = __builtin_amdgcn_rcpf(l[r]);   // l > 0 always (col 0 unmasked)
    const int rowg = wrow0 + r + ah;
#pragma unroll
    for (int j = 0; j < 4; ++j)
      Ob[(size_t)rowg * DHEAD + j * 16 + ln] = o[j][r] * inv;
  }
}

extern "C" void kernel_launch(void* const* d_in, const int* in_sizes, int n_in,
                              void* d_out, int out_size, void* d_ws, size_t ws_size,
                              hipStream_t stream) {
  const float* Q = (const float*)d_in[0];
  const float* K = (const float*)d_in[1];
  const float* V = (const float*)d_in[2];
  float* O = (float*)d_out;
  dim3 grid(SEQ / QT, BATCH);
  dim3 block(128);
  hipLaunchKernelGGL(fa_fwd_causal, grid, block, 0, stream, Q, K, V, O);
}
